// CLIP_36670430773295
// MI455X (gfx1250) — matrix operand order repaired and checked by reading, not compiled, not run on hardware
//
#include <hip/hip_runtime.h>

// ---------------------------------------------------------------------------
// CLIP KL loss, BS=4096, D=1024, on MI455X (gfx1250, wave32, WMMA).
// Algebra: label matrix == all ones -> q uniform -> log(BS) terms cancel:
//   loss = 1/4 * [ (1/BS) * sum(p1*logp1) - (1/BS^2) * sum(logp1)
//                + (1/BS) * sum(p2*logp2) - (1/BS^2) * sum(logp2) ]
//   p1 = row-softmax(scale*S), p2 = row-softmax(S^T),  S = i_n @ t_n^T.
// One bf16 WMMA GEMM with fused per-tile logsumexp stats; S never hits HBM.
// Double-buffered LDS staging fed by GLOBAL_LOAD_ASYNC_TO_LDS_B128
// (ASYNCcnt-tracked DMA, no VGPR round-trip), one barrier per K-step, async
// fill of the next tile overlapped with the 8 WMMAs of the current tile.
// ---------------------------------------------------------------------------

typedef __bf16 bf16_t;
typedef bf16_t v16bf __attribute__((ext_vector_type(16)));
typedef float  v8f   __attribute__((ext_vector_type(8)));
typedef int    v4i   __attribute__((vector_size(16)));   // matches builtin's pointee

#define BS    4096
#define DD    1024
#define TILE  128
#define TK    32
#define NITER (DD / TK)     // 32 K-steps
#define NKT   (BS / TILE)   // 32 tiles per output dimension
#define LDA   48            // LDS row stride (bf16 elems) for staged A/B (96B, 16B aligned)
#define LDS_S 132           // LDS row stride (floats) for S half-tile dump

#if __has_builtin(__builtin_amdgcn_global_load_async_to_lds_b128) && \
    __has_builtin(__builtin_amdgcn_s_wait_asynccnt)
#define USE_ASYNC 1
#else
#define USE_ASYNC 0
#endif

#if USE_ASYNC
typedef __attribute__((address_space(1))) v4i gv4i_t;
typedef __attribute__((address_space(3))) v4i lv4i_t;
__device__ __forceinline__ void async_b128(const unsigned short* g, unsigned short* l) {
  __builtin_amdgcn_global_load_async_to_lds_b128(
      (gv4i_t*)(uintptr_t)g, (lv4i_t*)(unsigned)(uintptr_t)l, 0, 0);
}
#endif

__device__ __forceinline__ unsigned short f2bf(float f) {
  union { float f; unsigned int u; } c; c.f = f;
  unsigned int u = c.u;
  u += 0x7fffu + ((u >> 16) & 1u);   // round to nearest even
  return (unsigned short)(u >> 16);
}

// --- Kernel 0: row-wise L2 normalize (eps 1e-12) + fp32 -> bf16 -------------
__global__ void __launch_bounds__(256)
k_normalize(const float* __restrict__ i_sh, const float* __restrict__ t_sh,
            unsigned short* __restrict__ iN, unsigned short* __restrict__ tN) {
  __shared__ float red[256];
  const int b   = blockIdx.x;           // 0..2*BS-1
  const int row = b & (BS - 1);
  const float*    src = (b < BS) ? i_sh : t_sh;
  unsigned short* dst = (b < BS) ? iN   : tN;
  const int tid = threadIdx.x;

  const float4 v = *(const float4*)(src + (size_t)row * DD + tid * 4);
  red[tid] = v.x * v.x + v.y * v.y + v.z * v.z + v.w * v.w;
  __syncthreads();
  for (int off = 128; off > 0; off >>= 1) {
    if (tid < off) red[tid] += red[tid + off];
    __syncthreads();
  }
  const float inv = 1.0f / fmaxf(sqrtf(red[0]), 1e-12f);
  ushort4 o;
  o.x = f2bf(v.x * inv); o.y = f2bf(v.y * inv);
  o.z = f2bf(v.z * inv); o.w = f2bf(v.w * inv);
  *(ushort4*)(dst + (size_t)row * DD + tid * 4) = o;
}

// --- Kernel 1: 128x128 S-tile via WMMA bf16 + fused per-tile softmax stats --
// Per-tile stats (float4): x=max(a), y=sum exp(a-max), z=sum a, w=sum a*exp(a-max)
// rows use a = scale*S, cols use a = S.
__global__ void __launch_bounds__(256)
k_gemm_stats(const unsigned short* __restrict__ iN,
             const unsigned short* __restrict__ tN,
             const float* __restrict__ scale_p,
             float4* __restrict__ rowStats,
             float4* __restrict__ colStats) {
  // [buf0 A][buf0 B][buf1 A][buf1 B], each TILE*LDA bf16 (12288 B) = 49152 B.
  // Reused as the 64x132 f32 S dump buffer (33792 B) in the epilogue.
  __shared__ __align__(16) unsigned char smem[4 * TILE * LDA * 2];
  unsigned short* stage = (unsigned short*)smem;
  float*          Sds   = (float*)smem;

  const int tid  = threadIdx.x;
  const int lane = tid & 31;
  const int w    = tid >> 5;    // wave 0..7
  const int wr   = w >> 1;      // 0..3: 32-row group
  const int wc   = w & 1;       // 0..1: 64-col group
  const int half = lane >> 4;   // 0/1
  const int l16  = lane & 15;

  const int nTile = blockIdx.x;
  const int mTile = blockIdx.y;
  const size_t mBase = (size_t)mTile * TILE;
  const size_t nBase = (size_t)nTile * TILE;

  // staging assignment: each thread moves 32B of A and 32B of B per fill
  const int sRow = tid >> 1;          // 0..127
  const int sSeg = (tid & 1) * 16;    // 0 or 16 (bf16 elems)
  const unsigned short* gArow = iN + (mBase + sRow) * DD + sSeg;
  const unsigned short* gBrow = tN + (nBase + sRow) * DD + sSeg;
  const int stageOff = sRow * LDA + sSeg;

  auto fill = [&](int buf, int kk) {
    const unsigned short* ga = gArow + kk;
    const unsigned short* gb = gBrow + kk;
    unsigned short* la = stage + buf * (2 * TILE * LDA) + stageOff;              // A
    unsigned short* lb = stage + buf * (2 * TILE * LDA) + TILE * LDA + stageOff; // B
#if USE_ASYNC
    async_b128(ga, la); async_b128(ga + 8, la + 8);
    async_b128(gb, lb); async_b128(gb + 8, lb + 8);
#else
    uint4 a0 = *(const uint4*)ga, a1 = *(const uint4*)(ga + 8);
    uint4 b0 = *(const uint4*)gb, b1 = *(const uint4*)(gb + 8);
    ((uint4*)la)[0] = a0; ((uint4*)la)[1] = a1;
    ((uint4*)lb)[0] = b0; ((uint4*)lb)[1] = b1;
#endif
  };

  v8f acc[2][4];
  v8f vz = {};
  #pragma unroll
  for (int mi = 0; mi < 2; ++mi)
    #pragma unroll
    for (int ni = 0; ni < 4; ++ni) acc[mi][ni] = vz;

  fill(0, 0);   // prologue: start buffer 0

  #pragma unroll 1
  for (int it = 0; it < NITER; ++it) {
    const int cur = it & 1;
#if USE_ASYNC
    __builtin_amdgcn_s_wait_asynccnt(0);   // my async writes to buf[cur] done
#endif
    __syncthreads();                        // everyone's writes done; prior reads of buf[cur^1] done
    if (it + 1 < NITER) fill(cur ^ 1, (it + 1) * TK);  // overlap with compute below

    const unsigned short* As = stage + cur * (2 * TILE * LDA);
    const unsigned short* Bs = As + TILE * LDA;

    // A frag: lane holds row (l16), 16 contiguous K from half*16
    v16bf af[2];
    #pragma unroll
    for (int mi = 0; mi < 2; ++mi) {
      const unsigned short* rp = As + (wr * 32 + mi * 16 + l16) * LDA;
      uint4* d = (uint4*)&af[mi];
      d[0] = *(const uint4*)(rp + half * 16);
      d[1] = *(const uint4*)(rp + half * 16 + 8);
    }
    // B frag: lane holds column (l16), 16 contiguous K from half*16
    v16bf bfr[4];
    #pragma unroll
    for (int ni = 0; ni < 4; ++ni) {
      const unsigned short* cp = Bs + (wc * 64 + ni * 16 + l16) * LDA + half * 16;
      uint4* d = (uint4*)&bfr[ni];
      d[0] = *(const uint4*)(cp);
      d[1] = *(const uint4*)(cp + 8);
    }
    #pragma unroll
    for (int mi = 0; mi < 2; ++mi)
      #pragma unroll
      for (int ni = 0; ni < 4; ++ni)
        acc[mi][ni] = __builtin_amdgcn_wmma_f32_16x16x32_bf16(
            false, af[mi], false, bfr[ni], (short)0, acc[mi][ni], false, false);
  }

  const float scaleV = scale_p[0];

  // per-tile stats, S dumped to LDS in two 64-row halves (reuses stage LDS)
  float cm = -3.0e38f, cz = 0.0f, ca = 0.0f, cw = 0.0f;  // col running stats
  #pragma unroll 1
  for (int ph = 0; ph < 2; ++ph) {
    __syncthreads();
    if ((wr >> 1) == ph) {
      // C layout: VGPR e -> M = e + 8*half (lanes split), N = l16
      #pragma unroll
      for (int mi = 0; mi < 2; ++mi)
        #pragma unroll
        for (int ni = 0; ni < 4; ++ni) {
          const int rloc = (wr & 1) * 32 + mi * 16 + half * 8;
          const int col  = wc * 64 + ni * 16 + l16;
          #pragma unroll
          for (int e = 0; e < 8; ++e)
            Sds[(rloc + e) * LDS_S + col] = acc[mi][ni][e];
        }
    }
    __syncthreads();
    if (tid < 64) {                       // row stats: a = scale*S
      const float* rp = Sds + tid * LDS_S;
      float m = -3.0e38f, A = 0.0f;
      for (int j = 0; j < TILE; ++j) { float a = scaleV * rp[j]; m = fmaxf(m, a); A += a; }
      float Z = 0.0f, W = 0.0f;
      for (int j = 0; j < TILE; ++j) {
        float a = scaleV * rp[j];
        float e = __expf(a - m);
        Z += e; W += a * e;
      }
      rowStats[(mBase + ph * 64 + tid) * NKT + nTile] = make_float4(m, Z, A, W);
    } else if (tid < 192) {               // col stats: a = S
      const int c = tid - 64;
      float pm = -3.0e38f, pA = 0.0f;
      for (int i = 0; i < 64; ++i) { float a = Sds[i * LDS_S + c]; pm = fmaxf(pm, a); pA += a; }
      float pZ = 0.0f, pW = 0.0f;
      for (int i = 0; i < 64; ++i) {
        float a = Sds[i * LDS_S + c];
        float e = __expf(a - pm);
        pZ += e; pW += a * e;
      }
      const float nm = fmaxf(cm, pm);
      const float s0 = __expf(cm - nm), s1 = __expf(pm - nm);
      cz = cz * s0 + pZ * s1;
      cw = cw * s0 + pW * s1;
      ca += pA; cm = nm;
    }
  }
  if (tid >= 64 && tid < 192)
    colStats[(nBase + (tid - 64)) * NKT + mTile] = make_float4(cm, cz, ca, cw);
}

// --- Kernel 2: merge 32 tile-stats per row/col, reduce to scalar (1 block) --
__global__ void __launch_bounds__(256)
k_final(const float4* __restrict__ rowStats, const float4* __restrict__ colStats,
        float* __restrict__ out) {
  __shared__ float red[256];
  const float invB = 1.0f / (float)BS;
  float local = 0.0f;
  for (int g = threadIdx.x; g < 2 * BS; g += 256) {
    const float4* st = (g < BS) ? (rowStats + (size_t)g * NKT)
                                : (colStats + (size_t)(g - BS) * NKT);
    float m = -3.0e38f;
    for (int t = 0; t < NKT; ++t) m = fmaxf(m, st[t].x);
    float Z = 0.0f, W = 0.0f, A = 0.0f;
    for (int t = 0; t < NKT; ++t) {
      float4 s = st[t];
      float sc = __expf(s.x - m);
      Z += s.y * sc; W += s.w * sc; A += s.z;
    }
    const float L  = m + __logf(Z);
    const float t1 = W / Z - L;            // sum p*logp over this row/col
    const float t2 = A - (float)BS * L;    // sum logp over this row/col
    local += 0.25f * (t1 * invB - t2 * invB * invB);
  }
  red[threadIdx.x] = local;
  __syncthreads();
  for (int off = 128; off > 0; off >>= 1) {
    if (threadIdx.x < off) red[threadIdx.x] += red[threadIdx.x + off];
    __syncthreads();
  }
  if (threadIdx.x == 0) out[0] = red[0];
}

extern "C" void kernel_launch(void* const* d_in, const int* in_sizes, int n_in,
                              void* d_out, int out_size, void* d_ws, size_t ws_size,
                              hipStream_t stream) {
  (void)in_sizes; (void)n_in; (void)out_size; (void)ws_size;
  const float* i_sh  = (const float*)d_in[0];
  const float* t_sh  = (const float*)d_in[1];
  const float* scale = (const float*)d_in[2];
  // d_in[3] (y) is mathematically dead: label collapses to all-ones.

  unsigned char* ws = (unsigned char*)d_ws;
  unsigned short* iN = (unsigned short*)(ws);                                  // 8 MB
  unsigned short* tN = (unsigned short*)(ws + (size_t)BS * DD * 2);            // 8 MB
  float4* rowStats   = (float4*)(ws + (size_t)BS * DD * 4);                    // 2 MB
  float4* colStats   = (float4*)(ws + (size_t)BS * DD * 4 + (size_t)BS * NKT * 16); // 2 MB

  k_normalize<<<2 * BS, 256, 0, stream>>>(i_sh, t_sh, iN, tN);
  dim3 g1(NKT, NKT);
  k_gemm_stats<<<g1, 256, 0, stream>>>(iN, tN, scale, rowStats, colStats);
  k_final<<<1, 256, 0, stream>>>(rowStats, colStats, (float*)d_out);
}
